// CUTSEncoder_884763263726
// MI455X (gfx1250) — matrix operand order repaired and checked by reading, not compiled, not run on hardware
//
#include <hip/hip_runtime.h>
#include <stdint.h>

typedef __attribute__((ext_vector_type(16))) _Float16 v16h;
typedef __attribute__((ext_vector_type(8)))  float    v8f;
typedef __attribute__((ext_vector_type(4)))  float    f4v;
typedef __attribute__((ext_vector_type(2)))  float    f2v;
typedef __attribute__((ext_vector_type(4)))  int      v4i;
typedef __attribute__((ext_vector_type(2)))  int      v2i;

// ---- gfx1250 async global->LDS copy (guarded; falls back to plain LDS stores) ----
#if defined(__HIP_DEVICE_COMPILE__) && \
    __has_builtin(__builtin_amdgcn_global_load_async_to_lds_b64) && \
    __has_builtin(__builtin_amdgcn_global_load_async_to_lds_b128)
#define USE_ASYNC_LDS 1
// pointer-to-AS1/AS3 typed pointees, per the builtin prototypes
typedef __attribute__((address_space(1))) v4i* gp_v4i;
typedef __attribute__((address_space(3))) v4i* lp_v4i;
typedef __attribute__((address_space(1))) v2i* gp_v2i;
typedef __attribute__((address_space(3))) v2i* lp_v2i;
#else
#define USE_ASYNC_LDS 0
#endif

__device__ __forceinline__ void wait_async0() {
#if __has_builtin(__builtin_amdgcn_s_wait_asynccnt)
  __builtin_amdgcn_s_wait_asynccnt(0);
#else
  asm volatile("s_wait_asynccnt 0" ::: "memory");
#endif
}

// ---------------- utility: zero a small float buffer ----------------
__global__ void k_zero(float* __restrict__ p, int n) {
  int i = blockIdx.x * blockDim.x + threadIdx.x;
  if (i < n) p[i] = 0.f;
}

// ---- pack conv weights (Cout,Cin,5,5) f32 -> WMMA B-fragment f16, (kt, ntile)-major ----
// wB[((kt*NTtot + ntile)*32 + lane)*16 + e]:
//   N = ntile*16 + (lane&15),  K = kt*32 + (lane>>4)*16 + e   (contiguous k order, same as A fragment)
__global__ void k_pack_w(const float* __restrict__ w, _Float16* __restrict__ wB,
                         int Cin, int Cout, int Kpad) {
  int idx = blockIdx.x * blockDim.x + threadIdx.x;
  int ktc = Kpad >> 5;
  int NTtot = Cout >> 4;
  int total = NTtot * ktc * 512;
  if (idx >= total) return;
  int e     = idx & 15;
  int l     = (idx >> 4) & 31;
  int slot  = idx >> 9;              // kt*NTtot + ntile
  int kt    = slot / NTtot;
  int ntile = slot - kt * NTtot;
  int n = (ntile << 4) + (l & 15);
  int k = (kt << 5) + ((l >> 4) << 4) + e;
  float v = 0.f;
  int KC = Cin * 25;
  if (k < KC) v = w[(n * Cin + k / 25) * 25 + (k % 25)];
  wB[idx] = (_Float16)v;
}

// ---------------- implicit-GEMM conv via WMMA ----------------
// One wave computes 16(M pixels) x (NT*16)(N out-channels) with NT accumulators.
// grid = (Mtiles/8, Cout/(16*NT)), block = 256 (8 waves, each its own M-tile;
// all waves share one async-staged, double-buffered B tile in LDS).
template <int NT>
__global__ __launch_bounds__(256)
void k_conv_wmma(const float* __restrict__ in, const _Float16* __restrict__ wB,
                 const float* __restrict__ bias, float* __restrict__ out,
                 float* __restrict__ stats, int Cin, int Cout, int Kpad) {
  const int lane   = threadIdx.x & 31;
  const int wave   = threadIdx.x >> 5;
  const int mtile  = blockIdx.x * 8 + wave;
  const int ntile0 = blockIdx.y * NT;
  const int NTtot  = Cout >> 4;
  const int KC     = Cin * 25;
  const int ktc    = Kpad >> 5;
  const int pixBase = mtile << 4;

  __shared__ __align__(32) _Float16 As[8][16][32];     // per-wave im2col staging (8 KB)
  __shared__ __align__(32) _Float16 Bs[2][NT * 512];   // double-buffered B tiles

  // per-lane fixed geometry: lane covers row m=lane>>1, K-halves kb..kb+15
  const int mrow = lane >> 1;
  const int kb   = (lane & 1) << 4;
  const int pix  = pixBase + mrow;
  const int bIdx = pix >> 16;
  const int rem  = pix & 65535;
  const int hPix = rem >> 8;
  const int wPix = rem & 255;
  const float* inB = in + (((size_t)bIdx * Cin) << 16);

  v8f acc[NT];
#pragma unroll
  for (int t = 0; t < NT; ++t) acc[t] = (v8f){};

  // stage B tile (NT KB, contiguous in packed buffer) into Bs[buf]
  auto stageB = [&](int kt, int buf) {
    const _Float16* g = wB + (size_t)(kt * NTtot + ntile0) * 512;
#if USE_ASYNC_LDS
    if (NT == 4) {
      __builtin_amdgcn_global_load_async_to_lds_b128(
          (gp_v4i)(g + threadIdx.x * 8), (lp_v4i)(&Bs[buf][threadIdx.x * 8]), 0, 0);
    } else {
      __builtin_amdgcn_global_load_async_to_lds_b64(
          (gp_v2i)(g + threadIdx.x * 4), (lp_v2i)(&Bs[buf][threadIdx.x * 4]), 0, 0);
    }
#else
    if (NT == 4) {
      *(f4v*)&Bs[buf][threadIdx.x * 8] = *(const f4v*)(g + threadIdx.x * 8);
    } else {
      *(f2v*)&Bs[buf][threadIdx.x * 4] = *(const f2v*)(g + threadIdx.x * 4);
    }
#endif
  };

  stageB(0, 0);

  for (int kt = 0; kt < ktc; ++kt) {
    // ---- A staging: gather 16 f16 into a register vector, one 32B LDS store ----
    const int k0 = (kt << 5) + kb;
    v16h av;
#pragma unroll
    for (int i = 0; i < 16; ++i) {
      int k = k0 + i;
      float v = 0.f;
      if (k < KC) {
        int ci = k / 25;
        int r  = k - ci * 25;
        int dy = r / 5;
        int dx = r - dy * 5;
        int hh = hPix + dy - 2; hh = hh < 0 ? 0 : (hh > 255 ? 255 : hh);
        int ww = wPix + dx - 2; ww = ww < 0 ? 0 : (ww > 255 ? 255 : ww);
        v = inB[(ci << 16) + (hh << 8) + ww];
      }
      av[i] = (_Float16)v;
    }
    *(v16h*)&As[wave][mrow][kb] = av;

#if USE_ASYNC_LDS
    wait_async0();                 // this wave's share of Bs[kt&1] landed
#endif
    __syncthreads();               // whole block: Bs[kt&1] + own-wave As visible

    if (kt + 1 < ktc) stageB(kt + 1, (kt + 1) & 1);   // overlap next B with compute

    // ---- A fragment: contiguous k order (k = 16*half + s), same as packed B ----
    v16h a;
    {
      int Mr = lane & 15, half = lane >> 4;
#pragma unroll
      for (int j = 0; j < 8; ++j) {
        int base = half * 16 + j * 2;
        a[2 * j]     = As[wave][Mr][base];
        a[2 * j + 1] = As[wave][Mr][base + 1];
      }
    }

    // ---- NT WMMAs reuse the same A fragment ----
    const _Float16* bbase = &Bs[kt & 1][0];
#pragma unroll
    for (int t = 0; t < NT; ++t) {
      v16h bf = *(const v16h*)(bbase + t * 512 + lane * 16);
      acc[t] = __builtin_amdgcn_wmma_f32_16x16x32_f16(
          false, a, false, bf, (short)0, acc[t], false, false);
    }
    __syncthreads();               // all reads of Bs[kt&1] done before it is re-staged
  }

  // ---- epilogue: bias add, store pre-BN y, accumulate BN stats ----
  const int nBase = (ntile0 << 4) + (lane & 15);
#pragma unroll
  for (int t = 0; t < NT; ++t) {
    int n = nBase + (t << 4);
    float bv = bias[n];
    float ss = 0.f, ss2 = 0.f;
#pragma unroll
    for (int r = 0; r < 8; ++r) {
      int m  = r + ((lane >> 4) << 3);   // C/D layout: VGPR r -> M=r (+8 hi half)
      int p  = pixBase + m;
      int bb = p >> 16;
      int rr = p & 65535;
      float y = acc[t][r] + bv;
      out[(((size_t)bb * Cout + n) << 16) + rr] = y;
      ss  += y;
      ss2 += y * y;
    }
    ss  += __shfl_down(ss, 16, 32);      // lanes l, l+16 share a channel
    ss2 += __shfl_down(ss2, 16, 32);
    if (lane < 16) {
      atomicAdd(&stats[2 * n],     ss);
      atomicAdd(&stats[2 * n + 1], ss2);
    }
  }
}

// ---------------- batch-norm apply + leaky-ReLU (in place) ----------------
__global__ void k_bn_lrelu(float* __restrict__ y, const float* __restrict__ g,
                           const float* __restrict__ be, const float* __restrict__ stats,
                           int Cmask) {
  size_t idx = (size_t)blockIdx.x * blockDim.x + threadIdx.x;
  int c = (int)((idx >> 16) & (unsigned)Cmask);
  const float invN = 1.f / 524288.f;            // B*H*W = 8*65536
  float mean = stats[2 * c] * invN;
  float var  = stats[2 * c + 1] * invN - mean * mean;
  float inv  = rsqrtf(var + 1e-5f);
  float v = (y[idx] - mean) * inv * g[c] + be[c];
  y[idx] = v >= 0.f ? v : 0.01f * v;
}

// ---------------- gather z at anchor / positive locations ----------------
__global__ void k_gather_z(const float* __restrict__ z, const int* __restrict__ anc,
                           const int* __restrict__ pos, float* __restrict__ za,
                           float* __restrict__ zp) {
  int idx = blockIdx.x * blockDim.x + threadIdx.x;   // over B*S*256
  int c  = idx & 255;
  int bs = idx >> 8;                                 // b*S + s
  int b  = bs >> 8;
  int ah = anc[2 * bs] + 2,  aw = anc[2 * bs + 1] + 2;
  int ph = pos[2 * bs] + 2,  pw = pos[2 * bs + 1] + 2;
  za[idx] = z[(((size_t)b * 256 + c) << 16) + (ah << 8) + aw];
  zp[idx] = z[(((size_t)b * 256 + c) << 16) + (ph << 8) + pw];
}

// ---------------- gather 5x5x3 real patches from x ----------------
__global__ void k_patch_real(const float* __restrict__ x, const int* __restrict__ anc,
                             float* __restrict__ pr) {
  int idx = blockIdx.x * blockDim.x + threadIdx.x;   // over B*S*3*5*5 = 153600
  if (idx >= 153600) return;
  int j  = idx % 5;
  int i  = (idx / 5) % 5;
  int c  = (idx / 25) % 3;
  int bs = idx / 75;
  int b  = bs >> 8;
  int ah = anc[2 * bs] + i;        // (anchor + 2) + (i - 2)
  int aw = anc[2 * bs + 1] + j;
  pr[idx] = x[(((size_t)b * 3 + c) << 16) + (ah << 8) + aw];
}

// ---------------- recon head: (B*S,256) @ (75,256)^T + b ----------------
__global__ void k_recon(const float* __restrict__ za, const float* __restrict__ rw,
                        const float* __restrict__ rb, float* __restrict__ out) {
  int idx = blockIdx.x * blockDim.x + threadIdx.x;   // over B*S*75
  if (idx >= 153600) return;
  int o  = idx % 75;
  int bs = idx / 75;
  const float* zr = za + (size_t)bs * 256;
  const float* wr = rw + (size_t)o * 256;
  float acc = rb[o];
#pragma unroll 8
  for (int k = 0; k < 256; ++k) acc = fmaf(zr[k], wr[k], acc);
  out[idx] = acc;
}

extern "C" void kernel_launch(void* const* d_in, const int* in_sizes, int n_in,
                              void* d_out, int out_size, void* d_ws, size_t ws_size,
                              hipStream_t stream) {
  const float* x   = (const float*)d_in[0];
  const int*   anc = (const int*)d_in[1];
  const int*   pos = (const int*)d_in[2];
  const float* w1  = (const float*)d_in[3];
  const float* b1  = (const float*)d_in[4];
  const float* g1  = (const float*)d_in[5];
  const float* be1 = (const float*)d_in[6];
  const float* w2  = (const float*)d_in[7];
  const float* b2  = (const float*)d_in[8];
  const float* g2  = (const float*)d_in[9];
  const float* be2 = (const float*)d_in[10];
  const float* w3  = (const float*)d_in[11];
  const float* b3  = (const float*)d_in[12];
  const float* g3  = (const float*)d_in[13];
  const float* be3 = (const float*)d_in[14];
  const float* w4  = (const float*)d_in[15];
  const float* b4  = (const float*)d_in[16];
  const float* g4  = (const float*)d_in[17];
  const float* be4 = (const float*)d_in[18];
  const float* rw  = (const float*)d_in[19];
  const float* rb  = (const float*)d_in[20];

  // ---- workspace layout ----
  char* ws = (char*)d_ws;
  float* z1 = (float*)ws;                           //  64 MB (8x32x256x256)
  float* z2 = (float*)(ws + (64ull  << 20));        // 128 MB
  float* z3 = (float*)(ws + (192ull << 20));        // 256 MB
  _Float16* w1p = (_Float16*)(ws + (448ull << 20)); // packed f16 weights
  _Float16* w2p = w1p + 2  * 3   * 512;             // L1: (32/16)*(96/32)*512
  _Float16* w3p = w2p + 4  * 25  * 512;             // L2
  _Float16* w4p = w3p + 8  * 50  * 512;             // L3
  float* stats  = (float*)(w4p + 16 * 100 * 512);   // L4
  float* s1 = stats;
  float* s2 = s1 + 2 * 32;
  float* s3 = s2 + 2 * 64;
  float* s4 = s3 + 2 * 128;

  // ---- output layout (flat tuple concat) ----
  float* oz  = (float*)d_out;                 // z: 8*256*256*256
  float* opr = oz  + 134217728ull;            // patch_real: 8*256*3*5*5
  float* orc = opr + 153600;                  // patch_recon
  float* oza = orc + 153600;                  // z_anchors: 8*256*256
  float* ozp = oza + 524288;                  // z_positives

  // ---- zero BN stat accumulators (atomically accumulated each call) ----
  k_zero<<<4, 256, 0, stream>>>(stats, 2 * (32 + 64 + 128 + 256));

  // ---- pack weights into WMMA B-fragment order, (kt, ntile)-major ----
  k_pack_w<<<(2  * 3   * 512 + 255) / 256, 256, 0, stream>>>(w1, w1p, 3,   32,  96);
  k_pack_w<<<(4  * 25  * 512 + 255) / 256, 256, 0, stream>>>(w2, w2p, 32,  64,  800);
  k_pack_w<<<(8  * 50  * 512 + 255) / 256, 256, 0, stream>>>(w3, w3p, 64,  128, 1600);
  k_pack_w<<<(16 * 100 * 512 + 255) / 256, 256, 0, stream>>>(w4, w4p, 128, 256, 3200);

  // ---- conv + BN chain (M-tiles = 524288/16 = 32768; 8 waves/block) ----
  dim3 cblk(256);
  k_conv_wmma<2><<<dim3(4096, 1), cblk, 0, stream>>>(x,  w1p, b1, z1, s1, 3,   32,  96);
  k_bn_lrelu<<<16777216  / 256, 256, 0, stream>>>(z1, g1, be1, s1, 31);
  k_conv_wmma<4><<<dim3(4096, 1), cblk, 0, stream>>>(z1, w2p, b2, z2, s2, 32,  64,  800);
  k_bn_lrelu<<<33554432  / 256, 256, 0, stream>>>(z2, g2, be2, s2, 63);
  k_conv_wmma<4><<<dim3(4096, 2), cblk, 0, stream>>>(z2, w3p, b3, z3, s3, 64,  128, 1600);
  k_bn_lrelu<<<67108864  / 256, 256, 0, stream>>>(z3, g3, be3, s3, 127);
  k_conv_wmma<4><<<dim3(4096, 4), cblk, 0, stream>>>(z3, w4p, b4, oz, s4, 128, 256, 3200);
  k_bn_lrelu<<<134217728 / 256, 256, 0, stream>>>(oz, g4, be4, s4, 255);

  // ---- gathers + recon head ----
  k_gather_z  <<<524288 / 256, 256, 0, stream>>>(oz, anc, pos, oza, ozp);
  k_patch_real<<<(153600 + 255) / 256, 256, 0, stream>>>(x, anc, opr);
  k_recon     <<<(153600 + 255) / 256, 256, 0, stream>>>(oza, rw, rb, orc);
}
